// MSAColumnAttention_46918222742021
// MI455X (gfx1250) — matrix operand order repaired and checked by reading, not compiled, not run on hardware
//
#include <hip/hip_runtime.h>
#include <hip/hip_bf16.h>
#include <math.h>

// ---- problem constants -----------------------------------------------------
#define SDIM 256          // MSA sequence length (attention axis)
#define RDIM 512          // residues
#define DDIM 256          // model dim
#define HNUM 8            // heads
#define CDIM 32           // head dim
#define NROWS (RDIM * SDIM)   // 131072 (r,s) rows
#define LN_EPS 1e-5f
#define QSCALE 0.17677669529663687f   // 1/sqrt(32)

typedef __attribute__((ext_vector_type(16))) __bf16 v16bf;
typedef __attribute__((ext_vector_type(8)))  __bf16 v8bf;
typedef __attribute__((ext_vector_type(8)))  float  v8f;

// ---- workspace layout (bytes) ---------------------------------------------
#define WS_WCAT 0
#define WS_MT   (size_t)(1u << 20)
#define WS_Q    (WS_MT + ((size_t)64 << 20))
#define WS_K    (WS_Q  + ((size_t)64 << 20))
#define WS_VT   (WS_K  + ((size_t)64 << 20))
#define WS_G    (WS_VT + ((size_t)64 << 20))

// ---- helpers ---------------------------------------------------------------
__device__ __forceinline__ v16bf cat8(v8bf lo, v8bf hi) {
  return __builtin_shufflevector(lo, hi, 0,1,2,3,4,5,6,7,8,9,10,11,12,13,14,15);
}
__device__ __forceinline__ v8f wmma_bf16(v16bf a, v16bf b, v8f c) {
  return __builtin_amdgcn_wmma_f32_16x16x32_bf16(false, a, false, b, (short)0, c,
                                                 false, false);
}
// A-operand 16x32 bf16 from row-major source: lane = row (lane&15),
// elems 0..15 -> k = 16*half + i
__device__ __forceinline__ v16bf load_a_tile(const __bf16* rowp, int half) {
  v8bf lo = *(const v8bf*)(rowp + 16 * half);
  v8bf hi = *(const v8bf*)(rowp + 16 * half + 8);
  return cat8(lo, hi);
}
// CDNA5 async global->LDS copy, 16B per lane; tracked by ASYNCcnt.
__device__ __forceinline__ void async_ld16(void* lds_dst, const void* gsrc) {
  unsigned lds_off = (unsigned)(unsigned long long)(uintptr_t)lds_dst;
  asm volatile("global_load_async_to_lds_b128 %0, %1, off"
               :: "v"(lds_off), "v"(gsrc) : "memory");
}
__device__ __forceinline__ void wait_asynccnt0() {
  asm volatile("s_wait_asynccnt 0x0" ::: "memory");
}

// ---- kernel 0: weights fp32 -> bf16 ---------------------------------------
__global__ __launch_bounds__(256) void k_wconv(
    const float* __restrict__ wq, const float* __restrict__ wk,
    const float* __restrict__ wv, const float* __restrict__ wg,
    const float* __restrict__ wo, __bf16* __restrict__ wcat) {
  int i = blockIdx.x * 256 + threadIdx.x;       // 0 .. 5*65536-1
  int mat = i >> 16, idx = i & 65535;
  float v;
  if      (mat == 0) v = wq[idx];
  else if (mat == 1) v = wk[idx];
  else if (mat == 2) v = wv[idx];
  else if (mat == 3) v = wg[idx];
  else               v = wo[idx];
  wcat[i] = (__bf16)v;
}

// ---- kernel 1: layernorm + [S,R,D]->[R,S,D] + bf16 ------------------------
__global__ __launch_bounds__(256) void k_layernorm(
    const float* __restrict__ M_raw, const float* __restrict__ sc,
    const float* __restrict__ bi, __bf16* __restrict__ Mt) {
  int wave = threadIdx.x >> 5, lane = threadIdx.x & 31;
  int row = blockIdx.x * 8 + wave;              // row = r*S + s
  int r = row >> 8, s = row & 255;
  const float* src = M_raw + ((size_t)s * RDIM + r) * DDIM + lane * 8;
  float4 x0 = ((const float4*)src)[0];
  float4 x1 = ((const float4*)src)[1];
  float xs[8] = {x0.x, x0.y, x0.z, x0.w, x1.x, x1.y, x1.z, x1.w};
  float sum = 0.f, sq = 0.f;
#pragma unroll
  for (int i = 0; i < 8; ++i) { sum += xs[i]; sq += xs[i] * xs[i]; }
#pragma unroll
  for (int m = 16; m >= 1; m >>= 1) {
    sum += __shfl_xor(sum, m, 32);
    sq  += __shfl_xor(sq,  m, 32);
  }
  float mu  = sum * (1.f / 256.f);
  float var = sq * (1.f / 256.f) - mu * mu;
  float rsd = rsqrtf(var + LN_EPS);
  int d = lane * 8;
  v8bf o;
#pragma unroll
  for (int i = 0; i < 8; ++i)
    o[i] = (__bf16)((xs[i] - mu) * rsd * sc[d + i] + bi[d + i]);
  *(v8bf*)(Mt + (size_t)row * DDIM + d) = o;
}

// ---- kernel 2: fused Q/K/V/gate projections, 2x2-blocked WMMA GEMM --------
// grid = (4096 row supertiles of 32, 4 col groups), block 256 (8 waves)
// wave col-tile pair ct0 = ((by*8+wave)*2) in [0,64): mat = ct0>>4
__global__ __launch_bounds__(256) void k_proj(
    const __bf16* __restrict__ Mt, const __bf16* __restrict__ wcat,
    const float* __restrict__ bg, __bf16* __restrict__ Q,
    __bf16* __restrict__ K, __bf16* __restrict__ Vt, __bf16* __restrict__ G) {
  __shared__ __bf16 sA[32 * 256];               // 16 KB, 32-row tile
  int rt = blockIdx.x;
  {
    const uint4* gsrc = (const uint4*)(Mt + (size_t)rt * 32 * DDIM);
    uint4* dst = (uint4*)sA;
#pragma unroll
    for (int i = 0; i < 4; ++i)
      dst[threadIdx.x + 256 * i] = gsrc[threadIdx.x + 256 * i];
  }
  __syncthreads();
  int wv = threadIdx.x >> 5, lane = threadIdx.x & 31;
  int half = lane >> 4, n = lane & 15;
  int ct0 = (blockIdx.y * 8 + wv) * 2;          // even, pair stays in one matrix
  int mat = ct0 >> 4;
  int e0  = (ct0 & 15) * 16 + n;
  int e1  = e0 + 16;
  const __bf16* Wb0 = wcat + ((size_t)mat * 256 + e0) * 256;
  const __bf16* Wb1 = wcat + ((size_t)mat * 256 + e1) * 256;
  v8f a00 = {}, a01 = {}, a10 = {}, a11 = {};
#pragma unroll
  for (int kb = 0; kb < 8; ++kb) {
    int k0 = kb * 32;
    __builtin_prefetch(Wb0 + k0 + 64, 0, 3);    // global_prefetch_b8
    v16bf A0 = load_a_tile(sA + (size_t)(lane & 15) * 256 + k0, half);
    v16bf A1 = load_a_tile(sA + (size_t)(16 + (lane & 15)) * 256 + k0, half);
    v16bf B0 = *(const v16bf*)(Wb0 + k0 + 16 * half);
    v16bf B1 = *(const v16bf*)(Wb1 + k0 + 16 * half);
    a00 = wmma_bf16(A0, B0, a00);
    a01 = wmma_bf16(A0, B1, a01);
    a10 = wmma_bf16(A1, B0, a10);
    a11 = wmma_bf16(A1, B1, a11);
  }
  int row0 = rt * 32;
  if (mat <= 1) {                               // Q (scaled) / K -> [r][h][s][c]
    __bf16* dst = (mat == 0) ? Q : K;
    float scl = (mat == 0) ? QSCALE : 1.f;
#pragma unroll
    for (int t = 0; t < 4; ++t) {               // t = rsub*2 + csub
      int rsub = (t >> 1) * 16, e = (t & 1) ? e1 : e0;
      const v8f& acc = (t == 0) ? a00 : (t == 1) ? a01 : (t == 2) ? a10 : a11;
      int h = e >> 5, c = e & 31;
#pragma unroll
      for (int j = 0; j < 8; ++j) {
        int row = row0 + rsub + 8 * half + j;
        int r = row >> 8, s = row & 255;
        dst[(((size_t)r * HNUM + h) * SDIM + s) * CDIM + c] =
            (__bf16)(acc[j] * scl);
      }
    }
  } else if (mat == 2) {                        // V transposed -> [r][h][c][s]
#pragma unroll
    for (int t = 0; t < 4; ++t) {
      int rsub = (t >> 1) * 16, e = (t & 1) ? e1 : e0;
      const v8f& acc = (t == 0) ? a00 : (t == 1) ? a01 : (t == 2) ? a10 : a11;
      int h = e >> 5, c = e & 31;
      int row = row0 + rsub + 8 * half;
      int r = row >> 8, s = row & 255;
      v8bf pk;
#pragma unroll
      for (int j = 0; j < 8; ++j) pk[j] = (__bf16)acc[j];
      *(v8bf*)(Vt + (((size_t)r * HNUM + h) * CDIM + c) * SDIM + s) = pk;
    }
  } else {                                      // gate = sigmoid(x + bg)
    float b0 = bg[e0], b1 = bg[e1];
#pragma unroll
    for (int t = 0; t < 4; ++t) {
      int rsub = (t >> 1) * 16, e = (t & 1) ? e1 : e0;
      float b = (t & 1) ? b1 : b0;
      const v8f& acc = (t == 0) ? a00 : (t == 1) ? a01 : (t == 2) ? a10 : a11;
#pragma unroll
      for (int j = 0; j < 8; ++j) {
        int row = row0 + rsub + 8 * half + j;
        G[(size_t)row * DDIM + e] = (__bf16)(1.f / (1.f + __expf(-(acc[j] + b))));
      }
    }
  }
}

// ---- kernel 3: per-(r,h) attention, flash-style online softmax ------------
// grid = (R, H), block 512 (16 waves); wave w owns query tile w (16 rows)
__global__ __launch_bounds__(512) void k_attn(
    const __bf16* __restrict__ Q, const __bf16* __restrict__ K,
    const __bf16* __restrict__ Vt, const __bf16* __restrict__ G,
    __bf16* __restrict__ GO) {
  __shared__ __bf16 sK[SDIM * CDIM];            // 16 KB  [s][c]
  __shared__ __bf16 sV[CDIM * SDIM];            // 16 KB  [c][s]
  __shared__ __bf16 sP[16][16 * 32];            // 16 KB  per-wave prob scratch
  int r = blockIdx.x, h = blockIdx.y;
  size_t rh = (size_t)r * HNUM + h;
  {
    // async global->LDS staging of K and V^T (ASYNCcnt path)
    const uint4* gk = (const uint4*)(K  + rh * SDIM * CDIM);
    const uint4* gv = (const uint4*)(Vt + rh * CDIM * SDIM);
    uint4* lk = (uint4*)sK;  uint4* lv = (uint4*)sV;
    async_ld16(lk + threadIdx.x,       gk + threadIdx.x);
    async_ld16(lk + threadIdx.x + 512, gk + threadIdx.x + 512);
    async_ld16(lv + threadIdx.x,       gv + threadIdx.x);
    async_ld16(lv + threadIdx.x + 512, gv + threadIdx.x + 512);
    wait_asynccnt0();
  }
  __syncthreads();
  int wv = threadIdx.x >> 5, lane = threadIdx.x & 31;
  int half = lane >> 4, n = lane & 15;
  int qt = wv;                                  // query tile
  v16bf qA = load_a_tile(Q + (rh * SDIM + qt * 16 + (lane & 15)) * CDIM, half);
  v8f o0 = {}, o1 = {};
  float mrun[8], lrun[8];
#pragma unroll
  for (int j = 0; j < 8; ++j) { mrun[j] = -INFINITY; lrun[j] = 0.f; }
#pragma unroll 1
  for (int kb = 0; kb < 8; ++kb) {
    int s0 = kb * 32;
    // logits: two 16-key tiles, K dim = C = 32 -> one WMMA each
    v16bf kt0 = *(const v16bf*)(sK + (size_t)(s0 + n) * CDIM + 16 * half);
    v16bf kt1 = *(const v16bf*)(sK + (size_t)(s0 + 16 + n) * CDIM + 16 * half);
    v8f z = {};
    v8f l0 = wmma_bf16(qA, kt0, z);
    v8f l1 = wmma_bf16(qA, kt1, z);
    // online softmax; row m = 8*half+j lives across the 16 lanes of a half
#pragma unroll
    for (int j = 0; j < 8; ++j) {
      float bm = fmaxf(l0[j], l1[j]);
      bm = fmaxf(bm, __shfl_xor(bm, 1, 32));
      bm = fmaxf(bm, __shfl_xor(bm, 2, 32));
      bm = fmaxf(bm, __shfl_xor(bm, 4, 32));
      bm = fmaxf(bm, __shfl_xor(bm, 8, 32));
      float mnew = fmaxf(mrun[j], bm);
      float corr = __expf(mrun[j] - mnew);
      float p0 = __expf(l0[j] - mnew);
      float p1 = __expf(l1[j] - mnew);
      float rs = p0 + p1;
      rs += __shfl_xor(rs, 1, 32);
      rs += __shfl_xor(rs, 2, 32);
      rs += __shfl_xor(rs, 4, 32);
      rs += __shfl_xor(rs, 8, 32);
      lrun[j] = lrun[j] * corr + rs;
      mrun[j] = mnew;
      o0[j] *= corr;  o1[j] *= corr;
      int m = 8 * half + j;
      sP[wv][m * 32 + n]      = (__bf16)p0;     // D-layout -> LDS
      sP[wv][m * 32 + 16 + n] = (__bf16)p1;
    }
    // re-read probs in A-layout (wave-private LDS; DS ops in-order per wave)
    v16bf pA = load_a_tile(&sP[wv][(lane & 15) * 32], half);
    // o += P(16x32) x V(32x32): two c-tiles, contiguous reads from Vt layout
    v16bf vt0 = *(const v16bf*)(sV + (size_t)n        * SDIM + s0 + 16 * half);
    v16bf vt1 = *(const v16bf*)(sV + (size_t)(16 + n) * SDIM + s0 + 16 * half);
    o0 = wmma_bf16(pA, vt0, o0);
    o1 = wmma_bf16(pA, vt1, o1);
  }
  // normalize, gate, store GO[row][e] bf16
  int e0 = h * CDIM;
#pragma unroll
  for (int j = 0; j < 8; ++j) {
    int s = qt * 16 + 8 * half + j;
    size_t base = ((size_t)r * SDIM + s) * DDIM + e0;
    float inv = 1.f / lrun[j];
    float g0 = (float)G[base + n];
    float g1 = (float)G[base + 16 + n];
    GO[base + n]      = (__bf16)(o0[j] * inv * g0);
    GO[base + 16 + n] = (__bf16)(o1[j] * inv * g1);
  }
}

// ---- kernel 4: output projection + bias + residual, back to [S,R,D] -------
// grid = (4096, 1), block 256 (8 waves); wave owns d-tile pair (wave*2)
__global__ __launch_bounds__(256) void k_outproj(
    const __bf16* __restrict__ GO, const __bf16* __restrict__ wcat,
    const float* __restrict__ bo, const float* __restrict__ M_raw,
    float* __restrict__ out) {
  __shared__ __bf16 sA[32 * 256];               // 16 KB, 32-row tile
  int rt = blockIdx.x;
  {
    const uint4* gsrc = (const uint4*)(GO + (size_t)rt * 32 * DDIM);
    uint4* dst = (uint4*)sA;
#pragma unroll
    for (int i = 0; i < 4; ++i)
      dst[threadIdx.x + 256 * i] = gsrc[threadIdx.x + 256 * i];
  }
  __syncthreads();
  int wv = threadIdx.x >> 5, lane = threadIdx.x & 31;
  int half = lane >> 4, n = lane & 15;
  int d0 = (wv * 2) * 16 + n;
  int d1 = d0 + 16;
  const __bf16* Wb0 = wcat + (size_t)4 * 65536 + (size_t)d0 * 256;
  const __bf16* Wb1 = wcat + (size_t)4 * 65536 + (size_t)d1 * 256;
  v8f a00 = {}, a01 = {}, a10 = {}, a11 = {};
#pragma unroll
  for (int kb = 0; kb < 8; ++kb) {
    int k0 = kb * 32;
    __builtin_prefetch(Wb0 + k0 + 64, 0, 3);
    v16bf A0 = load_a_tile(sA + (size_t)(lane & 15) * 256 + k0, half);
    v16bf A1 = load_a_tile(sA + (size_t)(16 + (lane & 15)) * 256 + k0, half);
    v16bf B0 = *(const v16bf*)(Wb0 + k0 + 16 * half);
    v16bf B1 = *(const v16bf*)(Wb1 + k0 + 16 * half);
    a00 = wmma_bf16(A0, B0, a00);
    a01 = wmma_bf16(A0, B1, a01);
    a10 = wmma_bf16(A1, B0, a10);
    a11 = wmma_bf16(A1, B1, a11);
  }
  float b0 = bo[d0], b1 = bo[d1];
#pragma unroll
  for (int t = 0; t < 4; ++t) {
    int rsub = (t >> 1) * 16;
    int d = (t & 1) ? d1 : d0;
    float bd = (t & 1) ? b1 : b0;
    const v8f& acc = (t == 0) ? a00 : (t == 1) ? a01 : (t == 2) ? a10 : a11;
#pragma unroll
    for (int j = 0; j < 8; ++j) {
      int row = rt * 32 + rsub + 8 * half + j;
      int r = row >> 8, s = row & 255;
      size_t off = ((size_t)s * RDIM + r) * DDIM + d;
      out[off] = M_raw[off] + acc[j] + bd;
    }
  }
}

// ---- host-side launcher ----------------------------------------------------
extern "C" void kernel_launch(void* const* d_in, const int* in_sizes, int n_in,
                              void* d_out, int out_size, void* d_ws, size_t ws_size,
                              hipStream_t stream) {
  const float* M_raw    = (const float*)d_in[0];
  const float* ln_scale = (const float*)d_in[1];
  const float* ln_bias  = (const float*)d_in[2];
  const float* Wq = (const float*)d_in[3];
  const float* Wk = (const float*)d_in[4];
  const float* Wv = (const float*)d_in[5];
  const float* Wg = (const float*)d_in[6];
  const float* bg = (const float*)d_in[7];
  const float* Wo = (const float*)d_in[8];
  const float* bo = (const float*)d_in[9];
  float* out = (float*)d_out;

  char* ws = (char*)d_ws;
  __bf16* wcat = (__bf16*)(ws + WS_WCAT);
  __bf16* Mt   = (__bf16*)(ws + WS_MT);   // reused as GO after k_proj
  __bf16* Qb   = (__bf16*)(ws + WS_Q);
  __bf16* Kb   = (__bf16*)(ws + WS_K);
  __bf16* Vtb  = (__bf16*)(ws + WS_VT);
  __bf16* Gb   = (__bf16*)(ws + WS_G);

  k_wconv<<<dim3(1280), dim3(256), 0, stream>>>(Wq, Wk, Wv, Wg, Wo, wcat);
  k_layernorm<<<dim3(NROWS / 8), dim3(256), 0, stream>>>(M_raw, ln_scale, ln_bias, Mt);
  k_proj<<<dim3(NROWS / 32, 4), dim3(256), 0, stream>>>(Mt, wcat, bg, Qb, Kb, Vtb, Gb);
  // Mt no longer needed -> reuse as gated attention output GO
  __bf16* GO = Mt;
  k_attn<<<dim3(RDIM, HNUM), dim3(512), 0, stream>>>(Qb, Kb, Vtb, Gb, GO);
  k_outproj<<<dim3(NROWS / 32, 1), dim3(256), 0, stream>>>(GO, wcat, bo, M_raw, out);
}
